// CMHSA_46943992545917
// MI455X (gfx1250) — matrix operand order repaired and checked by reading, not compiled, not run on hardware
//
#include <hip/hip_runtime.h>

#define B_   8
#define C_   256
#define T_   1024
#define NH_  8
#define HD_  32
#define EPS_ 1e-5f

typedef __attribute__((ext_vector_type(16))) __bf16   v16bf;
typedef __attribute__((ext_vector_type(8)))  float    v8f;
typedef __attribute__((ext_vector_type(8)))  unsigned v8u;

__device__ __forceinline__ __bf16 tobf(float f) { return (__bf16)f; }

__device__ __forceinline__ void split32(unsigned v, __bf16& lo, __bf16& hi) {
  lo = __builtin_bit_cast(__bf16, (unsigned short)(v & 0xffffu));
  hi = __builtin_bit_cast(__bf16, (unsigned short)(v >> 16));
}

__device__ __forceinline__ v8f wmma_bf16(v16bf a, v16bf b, v8f c) {
  // (neg_a, A, neg_b, B, c_mod, C, reuse_a, reuse_b)
  return __builtin_amdgcn_wmma_f32_16x16x32_bf16(false, a, false, b, (short)0, c,
                                                 false, false);
}

// Async global->LDS 128-bit copy (GLOBAL_LOAD_ASYNC_TO_LDS_B128, ASYNCcnt).
// lds_off: per-lane LDS byte address (low 32 bits of flat shared address).
// gaddr:   per-lane 64-bit global VA.
__device__ __forceinline__ void async_copy_b128(unsigned lds_off, const void* gaddr) {
  asm volatile("global_load_async_to_lds_b128 %0, %1, off"
               :
               : "v"(lds_off), "v"((unsigned long long)(uintptr_t)gaddr)
               : "memory");
}
__device__ __forceinline__ void wait_async0() {
  asm volatile("s_wait_asynccnt 0x0" ::: "memory");
}

// A fragment from a contiguous 16x32 row-major bf16 tile (K-pairs are dwords).
// lane m=l&15, half=l>>4; two contiguous 4-dword runs -> 2x ds_load_b128
__device__ __forceinline__ v16bf frag_a_u32(const __bf16* A, int lane) {
  const int m = lane & 15, hf4 = (lane >> 4) << 2;
  const unsigned* A32 = (const unsigned*)A + m * 16;
  v8u u;
#pragma unroll
  for (int j = 0; j < 8; ++j)
    u[j] = A32[((j < 4) ? j : j + 4) + hf4];
  return __builtin_bit_cast(v16bf, u);
}

// B fragment from a column-major-by-dword 32xN tile:
// dword (kp = k>>1, n) lives at dword index n*16 + kp.
// Lane (n, half) needs dwords n*16 + half*4 + (j<4 ? j : j+4), j=0..7 -> two 16B runs.
__device__ __forceinline__ v16bf frag_b_cm(const __bf16* Bs, int ncol, int lane) {
  const unsigned* B32 = (const unsigned*)Bs + ncol * 16 + ((lane >> 4) << 2);
  v8u u;
#pragma unroll
  for (int j = 0; j < 8; ++j) u[j] = B32[(j < 4) ? j : j + 4];
  return __builtin_bit_cast(v16bf, u);
}

// ---------------------------------------------------------------- cvt
__global__ void cvt_bf16(const float* __restrict__ src, __bf16* __restrict__ dst, int n) {
  int i = blockIdx.x * blockDim.x + threadIdx.x;
  const int stride = gridDim.x * blockDim.x;
  for (; i < n; i += stride) dst[i] = tobf(src[i]);
}

// ---------------------------------------------------------------- QKV GEMM
// q[b,o,t] = sum_c W[o,c] * x[b,c,t].  64x64 tile / WG, 8 waves each 16x32.
__global__ void __launch_bounds__(256) qkv_gemm(
    const __bf16* __restrict__ wq, const __bf16* __restrict__ wk,
    const __bf16* __restrict__ wv, const __bf16* __restrict__ xb,
    __bf16* __restrict__ qb, __bf16* __restrict__ kb, __bf16* __restrict__ vb) {
  __shared__ __align__(16) __bf16 As[64 * 32];
  __shared__ __align__(16) __bf16 Bs[32 * 64]; // column-major-by-dword
  const int tid = threadIdx.x, lane = tid & 31, wave = tid >> 5;
  const int n0 = blockIdx.x * 64, m0 = blockIdx.y * 64;
  const int mat = blockIdx.z >> 3, b = blockIdx.z & 7;
  const __bf16* A = (const __bf16*)__builtin_assume_aligned(
      (mat == 0) ? wq : (mat == 1) ? wk : wv, 256);
  __bf16* O = (__bf16*)__builtin_assume_aligned(
      (mat == 0) ? qb : (mat == 1) ? kb : vb, 256);
  const __bf16* Bg = (const __bf16*)__builtin_assume_aligned(xb, 256) +
                     (size_t)b * C_ * T_;
  const unsigned As_lds = (unsigned)(uintptr_t)&As[0];

  const int mloc = (wave >> 1) << 4, nloc = (wave & 1) << 5;
  v8f acc0 = {}, acc1 = {};
  for (int kk = 0; kk < C_; kk += 32) {
    // A tile: 64x32 row-major copy -> one async b128 per thread
    {
      const int r = tid >> 2, c4 = (tid & 3) * 4; // dword column
      async_copy_b128(As_lds + (unsigned)(tid * 16),
                      (const unsigned*)(A + (size_t)(m0 + r) * C_ + kk) + c4);
    }
    // B tile: 32x64 -> column-major-by-dword (bf16 transpose swizzle: VGPR path)
#pragma unroll
    for (int ii = 0; ii < 4; ++ii) {
      const int i32 = tid + ii * 256;
      const int r = i32 >> 5, c2 = i32 & 31; // k = r, n = 2*c2(+1)
      const unsigned v32 = *((const unsigned*)(Bg + (size_t)(kk + r) * T_ + n0) + c2);
      __bf16 lo, hi;
      split32(v32, lo, hi);
      const int kp = r >> 1, par = r & 1;
      Bs[((2 * c2) * 16 + kp) * 2 + par]     = lo;
      Bs[((2 * c2 + 1) * 16 + kp) * 2 + par] = hi;
    }
    if (kk + 32 < C_) {
      __builtin_prefetch((const void*)(A + (size_t)(m0 + (tid >> 2)) * C_ + kk + 32), 0, 3);
      __builtin_prefetch((const void*)(Bg + (size_t)(kk + 32 + (tid >> 3)) * T_ + n0), 0, 3);
    }
    wait_async0();
    __syncthreads();
    v16bf af = frag_a_u32(As + mloc * 32, lane);
    v16bf b0 = frag_b_cm(Bs, nloc + (lane & 15), lane);
    v16bf b1 = frag_b_cm(Bs, nloc + 16 + (lane & 15), lane);
    acc0 = wmma_bf16(af, b0, acc0);
    acc1 = wmma_bf16(af, b1, acc1);
    __syncthreads();
  }
  const int hf = lane >> 4, n = lane & 15;
#pragma unroll
  for (int j = 0; j < 8; ++j) {
    const int row = m0 + mloc + j + 8 * hf;
    const int col = n0 + nloc + n;
    O[((size_t)b * C_ + row) * T_ + col]      = tobf(acc0[j]);
    O[((size_t)b * C_ + row) * T_ + col + 16] = tobf(acc1[j]);
  }
}

// ---------------------------------------------------------------- V row sums
__global__ void vsum_kernel(const __bf16* __restrict__ vb, float* __restrict__ vsum) {
  const int idx = blockIdx.x * blockDim.x + threadIdx.x; // over B_*C_
  const unsigned* row =
      (const unsigned*)((const __bf16*)__builtin_assume_aligned(vb, 256) + (size_t)idx * T_);
  float s = 0.f;
#pragma unroll 8
  for (int t = 0; t < T_ / 2; ++t) {
    __bf16 lo, hi;
    split32(row[t], lo, hi);
    s += (float)lo + (float)hi;
  }
  vsum[idx] = s;
}

// ---------------------------------------------------------------- attention pass 1
// One WG per (b, 16 q-rows). Wave g computes raw scores for head g via WMMA,
// every wave h mixes all 8 raw tiles with w_head (head-interleaved LDS ->
// contiguous b128 reads), tracks online softmax stats per row; writes
// rowmax/rowsum, atomically accumulates sum(a^2) per (b,h). mean(a)==1/T.
__global__ void __launch_bounds__(256) attn_pass1(
    const __bf16* __restrict__ qb, const __bf16* __restrict__ kb,
    const float* __restrict__ w_head, float* __restrict__ rowmax,
    float* __restrict__ rowsum, float* __restrict__ sumsq) {
  __shared__ __align__(16) float sT[256 * NH_]; // [(m*16+n)*8 + g]
  __shared__ __align__(16) __bf16 Ksw[NH_ * 512]; // per-wave 32x16, col-major-by-dword
  const int lane = threadIdx.x, w = threadIdx.y;
  const int q0 = blockIdx.x * 16, b = blockIdx.y;
  const float scale = 0.17677669529663687f; // 32^-0.5
  const __bf16* qg =
      (const __bf16*)__builtin_assume_aligned(qb, 256) + ((size_t)b * C_ + w * HD_) * T_;
  const __bf16* kg =
      (const __bf16*)__builtin_assume_aligned(kb, 256) + ((size_t)b * C_ + w * HD_) * T_;
  __bf16* Kw = Ksw + w * 512;

  float whr[NH_];
#pragma unroll
  for (int g = 0; g < NH_; ++g) whr[g] = w_head[w * NH_ + g];

  const int hf = lane >> 4, n = lane & 15, kh = hf << 3;
  // A fragment = Q^T tile (A[m][k] = q[k][q0+m]); constant over the t loop
  v16bf af;
#pragma unroll
  for (int j = 0; j < 8; ++j) {
    const int k = ((j < 4) ? (2 * j) : (16 + 2 * (j - 4))) + kh;
    af[2 * j]     = qg[(size_t)k * T_ + q0 + n];
    af[2 * j + 1] = qg[(size_t)(k + 1) * T_ + q0 + n];
  }

  float mr[8], lr[8], ssr[8];
#pragma unroll
  for (int j = 0; j < 8; ++j) { mr[j] = -__builtin_inff(); lr[j] = 0.f; ssr[j] = 0.f; }

  for (int t0 = 0; t0 < T_; t0 += 16) {
    // stage this wave's 32(d) x 16(t) K tile, column-major-by-dword
#pragma unroll
    for (int i = 0; i < 8; ++i) {
      const int idx = (i << 5) + lane; // 0..255 dword positions
      const int k = idx >> 3, tp = idx & 7;
      const unsigned v32 = *((const unsigned*)(kg + (size_t)k * T_ + t0) + tp);
      __bf16 lo, hi;
      split32(v32, lo, hi);
      const int kp = k >> 1, par = k & 1;
      Kw[((2 * tp) * 16 + kp) * 2 + par]     = lo;
      Kw[((2 * tp + 1) * 16 + kp) * 2 + par] = hi;
    }
    if (t0 + 16 < T_)
      __builtin_prefetch((const void*)(kg + (size_t)lane * T_ + t0 + 16), 0, 3);
    v16bf bfr = frag_b_cm(Kw, n, lane);
    v8f sr = {};
    sr = wmma_bf16(af, bfr, sr);
#pragma unroll
    for (int j = 0; j < 8; ++j) sT[((j + 8 * hf) * 16 + n) * NH_ + w] = sr[j];
    __syncthreads();
#pragma unroll
    for (int j = 0; j < 8; ++j) {
      const float* srow = sT + ((j + 8 * hf) * 16 + n) * NH_;
      float v = 0.f;
#pragma unroll
      for (int g = 0; g < NH_; ++g) v += whr[g] * srow[g];
      v *= scale;
      const float mn = fmaxf(mr[j], v);
      const float e  = __expf(mr[j] - mn);
      lr[j]  = lr[j] * e + __expf(v - mn);
      ssr[j] = ssr[j] * e * e + __expf(2.f * (v - mn));
      mr[j]  = mn;
    }
    __syncthreads();
  }

  float ssacc = 0.f;
#pragma unroll
  for (int j = 0; j < 8; ++j) {
#pragma unroll
    for (int off = 8; off >= 1; off >>= 1) {
      const float mo = __shfl_xor(mr[j], off, 32);
      const float lo = __shfl_xor(lr[j], off, 32);
      const float so = __shfl_xor(ssr[j], off, 32);
      const float mn = fmaxf(mr[j], mo);
      lr[j]  = lr[j] * __expf(mr[j] - mn) + lo * __expf(mo - mn);
      ssr[j] = ssr[j] * __expf(2.f * (mr[j] - mn)) + so * __expf(2.f * (mo - mn));
      mr[j]  = mn;
    }
    if (n == 0) {
      const int m  = j + 8 * hf;
      const int bh = b * NH_ + w;
      rowmax[(size_t)bh * T_ + q0 + m] = mr[j];
      rowsum[(size_t)bh * T_ + q0 + m] = lr[j];
      ssacc += ssr[j] / (lr[j] * lr[j]);
    }
  }
  if (n == 0) atomicAdd(&sumsq[b * NH_ + w], ssacc);
}

// ---------------------------------------------------------------- attention pass 2
// Recompute mixed scores, build P=exp(s-M)/L as bf16 A-fragment via LDS,
// accumulate P x V^T with WMMA, apply affine norm, write (b,t,c) bf16.
__global__ void __launch_bounds__(256) attn_pass2(
    const __bf16* __restrict__ qb, const __bf16* __restrict__ kb,
    const __bf16* __restrict__ vb, const float* __restrict__ w_head,
    const float* __restrict__ gamma, const float* __restrict__ beta,
    const float* __restrict__ rowmax, const float* __restrict__ rowsum,
    const float* __restrict__ sumsq, const float* __restrict__ vsum,
    __bf16* __restrict__ attout) {
  __shared__ __align__(16) float sT[2 * 256 * NH_]; // [(tt*256 + m*16+n)*8 + g]
  __shared__ __align__(16) __bf16 pT[NH_ * 16 * 32];
  __shared__ __align__(16) __bf16 Ks2[NH_ * 1024]; // per-wave 32(d)x32(t), cm-by-dword
  const int lane = threadIdx.x, w = threadIdx.y;
  const int q0 = blockIdx.x * 16, b = blockIdx.y;
  const int bh = b * NH_ + w;
  const float scale = 0.17677669529663687f;
  const __bf16* qg =
      (const __bf16*)__builtin_assume_aligned(qb, 256) + ((size_t)b * C_ + w * HD_) * T_;
  const __bf16* kg =
      (const __bf16*)__builtin_assume_aligned(kb, 256) + ((size_t)b * C_ + w * HD_) * T_;
  const __bf16* vg =
      (const __bf16*)__builtin_assume_aligned(vb, 256) + ((size_t)b * C_ + w * HD_) * T_;
  __bf16* Kw = Ks2 + w * 1024;

  float whr[NH_];
#pragma unroll
  for (int g = 0; g < NH_; ++g) whr[g] = w_head[w * NH_ + g];

  const int hf = lane >> 4, n = lane & 15, kh = hf << 3, koff = hf << 3;
  float Mrow[8], invL[8];
#pragma unroll
  for (int j = 0; j < 8; ++j) {
    const int m = j + 8 * hf;
    Mrow[j] = rowmax[(size_t)bh * T_ + q0 + m];
    invL[j] = 1.f / rowsum[(size_t)bh * T_ + q0 + m];
  }

  v16bf af;
#pragma unroll
  for (int j = 0; j < 8; ++j) {
    const int k = ((j < 4) ? (2 * j) : (16 + 2 * (j - 4))) + kh;
    af[2 * j]     = qg[(size_t)k * T_ + q0 + n];
    af[2 * j + 1] = qg[(size_t)(k + 1) * T_ + q0 + n];
  }

  v8f acc0 = {}, acc1 = {};
  for (int t0 = 0; t0 < T_; t0 += 32) {
    // stage 32(d) x 32(t) K tile for this wave, column-major-by-dword
#pragma unroll
    for (int i = 0; i < 16; ++i) {
      const int idx = (i << 5) + lane; // 0..511 dword positions
      const int k = idx >> 4, tp = idx & 15;
      const unsigned v32 = *((const unsigned*)(kg + (size_t)k * T_ + t0) + tp);
      __bf16 lo, hi;
      split32(v32, lo, hi);
      const int kp = k >> 1, par = k & 1;
      Kw[((2 * tp) * 16 + kp) * 2 + par]     = lo;
      Kw[((2 * tp + 1) * 16 + kp) * 2 + par] = hi;
    }
    if (t0 + 32 < T_) {
      __builtin_prefetch((const void*)(kg + (size_t)lane * T_ + t0 + 32), 0, 3);
      __builtin_prefetch((const void*)(vg + (size_t)lane * T_ + t0 + 32), 0, 3);
    }
#pragma unroll
    for (int tt = 0; tt < 2; ++tt) {
      v16bf bfr = frag_b_cm(Kw, tt * 16 + n, lane);
      v8f sr = {};
      sr = wmma_bf16(af, bfr, sr);
#pragma unroll
      for (int j = 0; j < 8; ++j)
        sT[(tt * 256 + (j + 8 * hf) * 16 + n) * NH_ + w] = sr[j];
    }
    __syncthreads();
#pragma unroll
    for (int tt = 0; tt < 2; ++tt) {
#pragma unroll
      for (int j = 0; j < 8; ++j) {
        const int m = j + 8 * hf;
        const float* srow = sT + (tt * 256 + m * 16 + n) * NH_;
        float v = 0.f;
#pragma unroll
        for (int g = 0; g < NH_; ++g) v += whr[g] * srow[g];
        v *= scale;
        pT[w * 512 + m * 32 + tt * 16 + n] = tobf(__expf(v - Mrow[j]) * invL[j]);
      }
    }
    __syncthreads();
    // P (16x32) x V^T (32x32): B[k][d] = v[d][t0+k] -> two 4-dword runs per lane
    v16bf pa = frag_a_u32(pT + w * 512, lane);
    const unsigned* v0p = (const unsigned*)(vg + (size_t)n * T_ + t0 + koff);
    const unsigned* v1p = (const unsigned*)(vg + (size_t)(n + 16) * T_ + t0 + koff);
    v8u u0, u1;
#pragma unroll
    for (int j = 0; j < 8; ++j) { u0[j] = v0p[(j < 4) ? j : j + 4]; u1[j] = v1p[(j < 4) ? j : j + 4]; }
    acc0 = wmma_bf16(pa, __builtin_bit_cast(v16bf, u0), acc0);
    acc1 = wmma_bf16(pa, __builtin_bit_cast(v16bf, u1), acc1);
    __syncthreads();
  }

  // layer-norm of the attention map: mean == 1/T, var from sum(a^2)
  const float mean  = 1.f / (float)T_;
  const float var   = sumsq[bh] / ((float)T_ * (float)T_) - mean * mean;
  const float alpha = gamma[w] * rsqrtf(var + EPS_);
  const float cterm = beta[w] - alpha * mean;
#pragma unroll
  for (int j = 0; j < 8; ++j) {
    const int m = j + 8 * hf;
    const int q = q0 + m;
    const float o0 = alpha * acc0[j] + cterm * vsum[b * C_ + w * HD_ + n];
    const float o1 = alpha * acc1[j] + cterm * vsum[b * C_ + w * HD_ + 16 + n];
    attout[((size_t)b * T_ + q) * C_ + w * HD_ + n]      = tobf(o0);
    attout[((size_t)b * T_ + q) * C_ + w * HD_ + 16 + n] = tobf(o1);
  }
}

// ---------------------------------------------------------------- output projection
// y[b,o,t] = sum_c Wp[o,c] * attout[b,t,c] + bp[o]; written as (B,C,H,W) f32.
__global__ void __launch_bounds__(256) proj_gemm(
    const __bf16* __restrict__ wp, const __bf16* __restrict__ attout,
    const float* __restrict__ bp, float* __restrict__ out) {
  __shared__ __align__(16) __bf16 As[64 * 32];
  __shared__ __align__(16) __bf16 Bs[32 * 64]; // column-major-by-dword
  const int tid = threadIdx.x, lane = tid & 31, wave = tid >> 5;
  const int n0 = blockIdx.x * 64, m0 = blockIdx.y * 64;
  const int b = blockIdx.z;
  const __bf16* A  = (const __bf16*)__builtin_assume_aligned(wp, 256);
  const __bf16* Bg = (const __bf16*)__builtin_assume_aligned(attout, 256) +
                     (size_t)b * T_ * C_;
  const unsigned As_lds = (unsigned)(uintptr_t)&As[0];
  const unsigned Bs_lds = (unsigned)(uintptr_t)&Bs[0];

  const int mloc = (wave >> 1) << 4, nloc = (wave & 1) << 5;
  v8f acc0 = {}, acc1 = {};
  for (int kk = 0; kk < C_; kk += 32) {
    // A tile: 64x32 row-major copy -> one async b128 per thread
    {
      const int r = tid >> 2, c4 = (tid & 3) * 4;
      async_copy_b128(As_lds + (unsigned)(tid * 16),
                      (const unsigned*)(A + (size_t)(m0 + r) * C_ + kk) + c4);
    }
    // B[k][n] = attout[t0+n][k]: dword permutation, contiguous 4-dword runs both
    // sides -> one async b128 per thread into column-major-by-dword layout
    {
      const int c = tid >> 2, r4 = (tid & 3) * 4; // n = c, kp run = r4..r4+3
      async_copy_b128(Bs_lds + (unsigned)((c * 16 + r4) * 4),
                      (const unsigned*)(Bg + (size_t)(n0 + c) * C_ + kk) + r4);
    }
    if (kk + 32 < C_) {
      __builtin_prefetch((const void*)(A + (size_t)(m0 + (tid >> 2)) * C_ + kk + 32), 0, 3);
      __builtin_prefetch((const void*)(Bg + (size_t)(n0 + (tid >> 2)) * C_ + kk + 32), 0, 3);
    }
    wait_async0();
    __syncthreads();
    v16bf afr = frag_a_u32(As + mloc * 32, lane);
    v16bf b0  = frag_b_cm(Bs, nloc + (lane & 15), lane);
    v16bf b1  = frag_b_cm(Bs, nloc + 16 + (lane & 15), lane);
    acc0 = wmma_bf16(afr, b0, acc0);
    acc1 = wmma_bf16(afr, b1, acc1);
    __syncthreads();
  }
  const int hf = lane >> 4, n = lane & 15;
#pragma unroll
  for (int j = 0; j < 8; ++j) {
    const int row  = m0 + mloc + j + 8 * hf;
    const int col  = n0 + nloc + n;
    const float bb = bp[row];
    out[((size_t)b * C_ + row) * T_ + col]      = acc0[j] + bb;
    out[((size_t)b * C_ + row) * T_ + col + 16] = acc1[j] + bb;
  }
}

// ---------------------------------------------------------------- launch
extern "C" void kernel_launch(void* const* d_in, const int* in_sizes, int n_in,
                              void* d_out, int out_size, void* d_ws, size_t ws_size,
                              hipStream_t stream) {
  const float* x      = (const float*)d_in[0];
  const float* Wq     = (const float*)d_in[1];
  const float* Wk     = (const float*)d_in[2];
  const float* Wv     = (const float*)d_in[3];
  const float* w_head = (const float*)d_in[4];
  const float* gamma  = (const float*)d_in[5];
  const float* beta   = (const float*)d_in[6];
  const float* Wp     = (const float*)d_in[7];
  const float* bp     = (const float*)d_in[8];
  float* out = (float*)d_out;

  char* ws = (char*)d_ws;
  size_t off = 0;
  auto carve = [&](size_t bytes) -> void* {
    void* p = ws + off;
    off = (off + bytes + 255) & ~(size_t)255;
    return p;
  };
  const size_t NX = (size_t)B_ * C_ * T_;
  __bf16* xb   = (__bf16*)carve(NX * 2);
  __bf16* wqb  = (__bf16*)carve((size_t)C_ * C_ * 2);
  __bf16* wkb  = (__bf16*)carve((size_t)C_ * C_ * 2);
  __bf16* wvb  = (__bf16*)carve((size_t)C_ * C_ * 2);
  __bf16* wpb  = (__bf16*)carve((size_t)C_ * C_ * 2);
  __bf16* qbp  = (__bf16*)carve(NX * 2);
  __bf16* kbp  = (__bf16*)carve(NX * 2);
  __bf16* vbp  = (__bf16*)carve(NX * 2);
  __bf16* aout = (__bf16*)carve(NX * 2);
  float* rmax  = (float*)carve((size_t)B_ * NH_ * T_ * 4);
  float* rsum  = (float*)carve((size_t)B_ * NH_ * T_ * 4);
  float* ssq   = (float*)carve((size_t)B_ * NH_ * 4);
  float* vsum  = (float*)carve((size_t)B_ * C_ * 4);

  hipMemsetAsync(ssq, 0, (size_t)B_ * NH_ * sizeof(float), stream);

  cvt_bf16<<<512, 256, 0, stream>>>(x, xb, (int)NX);
  cvt_bf16<<<64, 256, 0, stream>>>(Wq, wqb, C_ * C_);
  cvt_bf16<<<64, 256, 0, stream>>>(Wk, wkb, C_ * C_);
  cvt_bf16<<<64, 256, 0, stream>>>(Wv, wvb, C_ * C_);
  cvt_bf16<<<64, 256, 0, stream>>>(Wp, wpb, C_ * C_);

  qkv_gemm<<<dim3(T_ / 64, C_ / 64, 3 * B_), 256, 0, stream>>>(
      wqb, wkb, wvb, xb, qbp, kbp, vbp);

  vsum_kernel<<<dim3((B_ * C_) / 256), 256, 0, stream>>>(vbp, vsum);

  attn_pass1<<<dim3(T_ / 16, B_), dim3(32, 8), 0, stream>>>(
      qbp, kbp, w_head, rmax, rsum, ssq);

  attn_pass2<<<dim3(T_ / 16, B_), dim3(32, 8), 0, stream>>>(
      qbp, kbp, vbp, w_head, gamma, beta, rmax, rsum, ssq, vsum, aout);

  proj_gemm<<<dim3(T_ / 64, C_ / 64, B_), 256, 0, stream>>>(wpb, aout, bp, out);

  (void)in_sizes; (void)n_in; (void)out_size; (void)ws_size;
}
